// SobelLossMean_24558622998861
// MI455X (gfx1250) — compile-verified, matrix operand order checked by reading, not run on hardware
//
#include <hip/hip_runtime.h>

// ---------------------------------------------------------------------------
// SobelLossMean on MI455X (gfx1250, wave32)
//
// Roofline: 201 MB read, scalar out -> HBM-bound (~8.6 us @ 23.3 TB/s).
// Strategy: single pass, gray computed once per pixel into LDS halo tiles,
// exact integer accumulation (deterministic u32 atomics), WMMA-based wave32
// reduction (exact: all values are small integers).
// ---------------------------------------------------------------------------

typedef __attribute__((ext_vector_type(16))) _Float16 v16h;
typedef __attribute__((ext_vector_type(8)))  float    v8f;

#define IMG_H 512
#define IMG_W 512
#define HW    (IMG_H * IMG_W)
#define CHW   (3 * HW)

#define TX 64           // output tile width
#define TY 16           // output tile height
#define HX (TX + 2)     // halo tile width
#define HY (TY + 2)     // halo tile height
#define NTHREADS 256    // 8 wave32s

// mul(255).clamp(0,255) -> uint8 truncation (values already >= 0)
__device__ __forceinline__ float q255(float v) {
    return floorf(fminf(fmaxf(v * 255.0f, 0.0f), 255.0f));
}

// cv2 RGB2GRAY port: round(0.299 R + 0.587 G + 0.114 B), round-half-even
__device__ __forceinline__ float gray_at(const float* __restrict__ p, int idx) {
    float r = q255(p[idx]);
    float g = q255(p[idx + HW]);
    float b = q255(p[idx + 2 * HW]);
    return rintf(0.299f * r + 0.587f * g + 0.114f * b);
}

__global__ void sobel_loss_init(unsigned int* acc) { *acc = 0u; }

__global__ __launch_bounds__(NTHREADS)
void sobel_loss_main(const float* __restrict__ pred,
                     const float* __restrict__ tru,
                     unsigned int* __restrict__ acc)
{
    __shared__ float gp[HY][HX];             // gray(pred) with 1-px halo
    __shared__ float gt[HY][HX];             // gray(true) with 1-px halo
    __shared__ float wpart[NTHREADS / 32];   // per-wave partials

    const int tid = threadIdx.x;
    const int bx  = blockIdx.x * TX;
    const int by  = blockIdx.y * TY;
    const long img = blockIdx.z;
    const float* pb = pred + img * (long)CHW;
    const float* tb = tru  + img * (long)CHW;

    // ---- Stage grays (with BORDER_REFLECT_101 halo) into LDS -------------
    for (int i = tid; i < HX * HY; i += NTHREADS) {
        int ly = i / HX, lx = i % HX;
        int gy = by - 1 + ly;
        int gx = bx - 1 + lx;
        // halo is only ever 1 out of range: reflect-101 is -1->1, N->N-2
        gy = (gy < 0) ? 1 : ((gy > IMG_H - 1) ? (IMG_H - 2) : gy);
        gx = (gx < 0) ? 1 : ((gx > IMG_W - 1) ? (IMG_W - 2) : gx);
        int idx = gy * IMG_W + gx;
        gp[ly][lx] = gray_at(pb, idx);
        gt[ly][lx] = gray_at(tb, idx);
    }
    __syncthreads();

    // ---- Sobel ksize=1 + convertScaleAbs + addWeighted, |p - t| ----------
    float lsum = 0.0f;   // integer-valued; <= 4 * 255 = 1020
    for (int i = tid; i < TX * TY; i += NTHREADS) {
        int oy = i / TX + 1, ox = i % TX + 1;

        float psx = fminf(fabsf(gp[oy][ox + 1] - gp[oy][ox - 1]), 255.0f);
        float psy = fminf(fabsf(gp[oy + 1][ox] - gp[oy - 1][ox]), 255.0f);
        float pv  = fminf(rintf(0.5f * psx + 0.5f * psy), 255.0f);

        float tsx = fminf(fabsf(gt[oy][ox + 1] - gt[oy][ox - 1]), 255.0f);
        float tsy = fminf(fabsf(gt[oy + 1][ox] - gt[oy - 1][ox]), 255.0f);
        float tv  = fminf(rintf(0.5f * tsx + 0.5f * tsy), 255.0f);

        lsum += fabsf(pv - tv);   // integer in [0,255]
    }

    // ---- wave32 reduction on the matrix unit -----------------------------
    // A(16x32 f16): lane L contributes lsum at (row L%16, K = 0 or 8).
    // B = all-ones => D[m][n] = rowsum_m (replicated over n). Summing a
    // lane's 8 accumulator VGPRs gives sum(rows 0-7) for lanes 0-15 and
    // sum(rows 8-15) for lanes 16-31; one xor-16 shuffle combines them.
    // All values are integers < 2^21: f16 inputs and f32 math are exact.
    v16h a = {};
    a[0] = (_Float16)lsum;
    v16h ones;
    #pragma unroll
    for (int c = 0; c < 16; ++c) ones[c] = (_Float16)1.0f;
    v8f cacc = {};
    cacc = __builtin_amdgcn_wmma_f32_16x16x32_f16(
        /*neg_a=*/false, a, /*neg_b=*/false, ones,
        /*c_mod=*/(short)0, cacc, /*reuse_a=*/false, /*reuse_b=*/false);
    float wsum = 0.0f;
    #pragma unroll
    for (int c = 0; c < 8; ++c) wsum += cacc[c];
    wsum += __shfl_xor(wsum, 16, 32);

    if ((tid & 31) == 0) wpart[tid >> 5] = wsum;
    __syncthreads();
    if (tid == 0) {
        float tot = 0.0f;
        #pragma unroll
        for (int w = 0; w < NTHREADS / 32; ++w) tot += wpart[w];
        // tot is an exact integer <= 261120; u32 atomics are deterministic
        atomicAdd(acc, (unsigned int)tot);
    }
}

__global__ void sobel_loss_final(const unsigned int* __restrict__ acc,
                                 float* __restrict__ out, double denom)
{
    *out = (float)((double)(*acc) / denom);
}

extern "C" void kernel_launch(void* const* d_in, const int* in_sizes, int n_in,
                              void* d_out, int out_size, void* d_ws, size_t ws_size,
                              hipStream_t stream)
{
    const float* pred = (const float*)d_in[0];
    const float* tru  = (const float*)d_in[1];
    unsigned int* acc = (unsigned int*)d_ws;

    const int B = in_sizes[0] / CHW;   // 32 for the reference setup

    sobel_loss_init<<<1, 1, 0, stream>>>(acc);

    dim3 grid(IMG_W / TX, IMG_H / TY, B);
    sobel_loss_main<<<grid, NTHREADS, 0, stream>>>(pred, tru, acc);

    sobel_loss_final<<<1, 1, 0, stream>>>(
        acc, (float*)d_out, 255.0 * (double)B * (double)HW);
}
